// Model_68023692034061
// MI455X (gfx1250) — matrix operand order repaired and checked by reading, not compiled, not run on hardware
//
#include <hip/hip_runtime.h>

#define DEV __device__ __forceinline__

typedef __attribute__((ext_vector_type(16))) _Float16 v16h;
typedef __attribute__((ext_vector_type(8)))  _Float16 v8h;
typedef __attribute__((ext_vector_type(8)))  float    v8f;

constexpr int BB = 4, TT = 2048, DD = 2048, D2 = 1024, CC = 20;
constexpr int TP = TT + 2;   // padded rows (1 zero halo row on each side)

DEV v16h cat16(v8h lo, v8h hi) {
  return __builtin_shufflevector(lo, hi, 0,1,2,3,4,5,6,7,8,9,10,11,12,13,14,15);
}
DEV v8f zero8() { v8f z;
#pragma unroll
  for (int i = 0; i < 8; ++i) z[i] = 0.0f;
  return z;
}

// A-fragment 16x32 f16: lane m (=lane&15) holds row M=m; K = hi*8.. and 16+hi*8..
DEV v16h load_A(const _Float16* p, int ld, int lane) {
  int m = lane & 15, hi = lane >> 4;
  const _Float16* q = p + (size_t)m * ld + hi * 8;
  return cat16(*(const v8h*)q, *(const v8h*)(q + 16));
}
// B-fragment 32x16 f16: lane holds column N=lane&15; K = hi*8.. and 16+hi*8..
DEV v16h load_B(const _Float16* p, int ld, int lane) {
  int m = lane & 15, hi = lane >> 4;
  const _Float16* q = p + (size_t)m * ld + hi * 8;
  return cat16(*(const v8h*)q, *(const v8h*)(q + 16));
}
DEV v8f wmma16(v16h a, v16h b, v8f c) {
  return __builtin_amdgcn_wmma_f32_16x16x32_f16(false, a, false, b, (short)0, c, false, false);
}

DEV float wave_sum(float v) {
  for (int o = 16; o; o >>= 1) v += __shfl_xor(v, o, 32);
  return v;
}
DEV float wave_max(float v) {
  for (int o = 16; o; o >>= 1) v = fmaxf(v, __shfl_xor(v, o, 32));
  return v;
}
// wave-private LDS store->load ordering (no block barrier needed)
DEV void lds_fence() {
  __builtin_amdgcn_wave_barrier();
  asm volatile("s_wait_dscnt 0" ::: "memory");
  __builtin_amdgcn_wave_barrier();
}
// async copy 16B global -> LDS (per lane), tracked by ASYNCcnt
DEV void async_cp16(_Float16* lds_dst, const _Float16* gsrc) {
  unsigned off = (unsigned)(uintptr_t)lds_dst;   // generic LDS addr low 32b = wave-relative offset
  asm volatile("global_load_async_to_lds_b128 %0, %1, off"
               :: "v"(off), "v"(gsrc) : "memory");
}
DEV void wait_async0() { asm volatile("s_wait_asynccnt 0" ::: "memory"); }

// ---------------- small prep kernels ----------------

// zero the halo rows of both padded activation buffers
__global__ void k_zeropad(_Float16* a, _Float16* bb) {
  int r = blockIdx.x;                     // 16 halo rows total
  _Float16* base = (r < 8) ? a : bb;
  int rr = r & 7;
  int b = rr >> 1, top = rr & 1;
  _Float16* row = base + ((size_t)b * TP + (top ? (TT + 1) : 0)) * DD;
  for (int c = threadIdx.x; c < DD; c += blockDim.x) row[c] = (_Float16)0.0f;
}

// x (B,T,D) f32 -> padded (B,T+2,D) f16 interior
__global__ void k_cvt(const float* __restrict__ x, _Float16* __restrict__ xh) {
  int r = blockIdx.x;                     // 8192 = B*T rows
  int b = r >> 11, t = r & 2047;
  const float* src = x + (size_t)r * DD;
  _Float16* dst = xh + ((size_t)b * TP + t + 1) * DD;
  for (int c = threadIdx.x; c < DD; c += blockDim.x) dst[c] = (_Float16)src[c];
}

// w (D, D/2, 3) f32 -> wp[o][kk*1024+i] f16 (GEMM B operand, K contiguous per out-channel)
__global__ void k_prepack(const float* __restrict__ w1, const float* __restrict__ w2,
                          _Float16* __restrict__ wp1, _Float16* __restrict__ wp2) {
  const size_t NW = (size_t)DD * 3072;
  for (size_t idx = (size_t)blockIdx.x * blockDim.x + threadIdx.x; idx < 2 * NW;
       idx += (size_t)gridDim.x * blockDim.x) {
    int which = idx >= NW;
    size_t e = which ? idx - NW : idx;
    size_t o = e / 3072;
    int r = (int)(e % 3072);
    int kk = r >> 10, i = r & 1023;
    const float* src = which ? w2 : w1;
    _Float16* dst = which ? wp2 : wp1;
    dst[e] = (_Float16)src[o * 3072 + (size_t)i * 3 + kk];
  }
}

// proxy (1, C, D) -> pn[br][c][k] f32, L2-normalized over the 1024-wide half
__global__ void k_proxy(const float* __restrict__ proxy, float* __restrict__ pn) {
  int w = blockIdx.x;                     // 40 = 2 branches * 20 classes
  int br = w / CC, c = w % CC;
  int lane = threadIdx.x;
  const float* src = proxy + (size_t)c * DD + br * D2;
  float ss = 0.f;
  for (int k = lane; k < D2; k += 32) ss += src[k] * src[k];
  ss = wave_sum(ss);
  float inv = 1.0f / fmaxf(sqrtf(ss), 1e-12f);
  float* dst = pn + ((size_t)br * CC + c) * D2;
  for (int k = lane; k < D2; k += 32) dst[k] = src[k] * inv;
}

// ---------------- grouped conv1d (k=3, groups=2) as implicit GEMM ----------------
// in/out: padded (B, T+2, D) f16. Block = 4 waves; each wave: 32 t-rows x 64 out-chan.
// Weight chunk (64 cols x 32 K) staged cooperatively via async-to-LDS, double-buffered,
// branch-free (per-thread staging pointers precomputed; chunk offset is just 32*c).
__global__ void __launch_bounds__(128) k_conv(const _Float16* __restrict__ in,
                                              const _Float16* __restrict__ wp,
                                              const float* __restrict__ bias,
                                              _Float16* __restrict__ out) {
  __shared__ __align__(16) _Float16 wt[2][64 * 40];   // stride 40 to spread LDS banks
  int tid = threadIdx.x;
  int lane = tid & 31, warp = tid >> 5;
  int blk = blockIdx.x;                 // 2048 = 4b * 16tq * 32oq
  int oq = blk & 31;
  int tq = (blk >> 5) & 15;
  int b = blk >> 9;
  int o0 = oq << 6;
  int g = o0 >> 10;
  int tbase = tq * 128 + warp * 32;
  int m = lane & 15, hi = lane >> 4;
  const _Float16* in_b = in + ((size_t)b * TP + 1) * DD + g * D2;  // logical row 0
  const _Float16* wbase = wp + (size_t)o0 * 3072;

  // staging thread-constants: each thread copies 2 x 16B per chunk (cols j and j+32)
  int j_a = tid >> 2, kp_a = tid & 3;
  const _Float16* wsrc0 = wbase + (size_t)j_a * 3072 + kp_a * 8;
  const _Float16* wsrc1 = wbase + (size_t)(j_a + 32) * 3072 + kp_a * 8;
  _Float16* wdst0 = &wt[0][j_a * 40 + kp_a * 8];
  _Float16* wdst1 = wdst0 + 32 * 40;
  auto stage = [&](int c, int buf) {    // chunk c: weight offset == 32*c (kk*1024+i0)
    const _Float16* s0 = wsrc0 + (size_t)32 * c;
    const _Float16* s1 = wsrc1 + (size_t)32 * c;
    async_cp16(wdst0 + buf * (64 * 40), s0);
    async_cp16(wdst1 + buf * (64 * 40), s1);
  };

  v8f acc[8];
#pragma unroll
  for (int i = 0; i < 8; ++i) acc[i] = zero8();

  stage(0, 0);
  wait_async0();
  __syncthreads();

  int c = 0;
  for (int kk = 0; kk < 3; ++kk) {
    const _Float16* arow_k = in_b + (size_t)(tbase + kk - 1) * DD;
    for (int ic = 0; ic < 32; ++ic, ++c) {
      int cur = c & 1;
      if (c < 95) stage(c + 1, cur ^ 1);
      const _Float16* arow = arow_k + ic * 32;
      __builtin_prefetch(arow + (size_t)m * DD + 256, 0, 3);   // WGP-scope prefetch
      v16h a0 = load_A(arow, DD, lane);
      v16h a1 = load_A(arow + (size_t)16 * DD, DD, lane);
      v16h b0 = load_B(&wt[cur][0 * 40], 40, lane);
      v16h b1 = load_B(&wt[cur][16 * 40], 40, lane);
      v16h b2 = load_B(&wt[cur][32 * 40], 40, lane);
      v16h b3 = load_B(&wt[cur][48 * 40], 40, lane);
      acc[0] = wmma16(a0, b0, acc[0]);
      acc[1] = wmma16(a0, b1, acc[1]);
      acc[2] = wmma16(a0, b2, acc[2]);
      acc[3] = wmma16(a0, b3, acc[3]);
      acc[4] = wmma16(a1, b0, acc[4]);
      acc[5] = wmma16(a1, b1, acc[5]);
      acc[6] = wmma16(a1, b2, acc[6]);
      acc[7] = wmma16(a1, b3, acc[7]);
      if (c < 95) {
        wait_async0();          // own wave's staging of next buffer done
        __syncthreads();        // all waves staged + finished reading old buffer
      }
    }
  }
#pragma unroll
  for (int sub = 0; sub < 2; ++sub)
#pragma unroll
    for (int n = 0; n < 4; ++n) {
      int oc = o0 + n * 16 + m;
      float bv = bias[oc];
      v8f av = acc[sub * 4 + n];
#pragma unroll
      for (int r = 0; r < 8; ++r) {
        int t = tbase + sub * 16 + r + 8 * hi;
        float v = fmaxf(av[r] + bv, 0.0f);
        out[((size_t)b * TP + t + 1) * DD + oc] = (_Float16)v;
      }
    }
}

// ---------------- per-(b,t): L2-normalize halves, proxy softmax scores ----------------
__global__ void __launch_bounds__(128) k_norm_feat(const _Float16* __restrict__ h2,
                                                   const float* __restrict__ pn,
                                                   _Float16* __restrict__ onorm,
                                                   _Float16* __restrict__ scoreT) {
  __shared__ __align__(16) _Float16 sn[4][2048];
  int lane = threadIdx.x & 31, warp = threadIdx.x >> 5;
  int w = blockIdx.x * 4 + warp;   // 8192 waves = B*T
  int b = w >> 11;
  int t = w & 2047;
  const _Float16* row = h2 + ((size_t)b * TP + t + 1) * DD;
  _Float16* orow = onorm + ((size_t)b * TP + t + 1) * DD;
  float ss0 = 0.f, ss1 = 0.f;
  for (int g8 = lane; g8 < 256; g8 += 32) {
    v8h v = *(const v8h*)(row + g8 * 8);
    float s = 0.f;
#pragma unroll
    for (int i = 0; i < 8; ++i) { float f = (float)v[i]; s += f * f; }
    if (g8 < 128) ss0 += s; else ss1 += s;
  }
  ss0 = wave_sum(ss0);
  ss1 = wave_sum(ss1);
  float inv0 = 1.0f / fmaxf(sqrtf(ss0), 1e-12f);
  float inv1 = 1.0f / fmaxf(sqrtf(ss1), 1e-12f);
  for (int g8 = lane; g8 < 256; g8 += 32) {
    float inv = (g8 < 128) ? inv0 : inv1;
    v8h v = *(const v8h*)(row + g8 * 8);
    v8h nv;
#pragma unroll
    for (int i = 0; i < 8; ++i) nv[i] = (_Float16)((float)v[i] * inv);
    *(v8h*)(orow + g8 * 8) = nv;
    *(v8h*)(&sn[warp][g8 * 8]) = nv;
  }
  lds_fence();
  int c = lane < CC ? lane : CC - 1;
  for (int br = 0; br < 2; ++br) {
    const float* pr = pn + ((size_t)br * CC + c) * D2;
    const _Float16* ov = &sn[warp][br * D2];
    float acc = 0.f;
    for (int k8 = 0; k8 < 128; ++k8) {
      v8h o8 = *(const v8h*)(ov + k8 * 8);
      v8f p8 = *(const v8f*)(pr + k8 * 8);
#pragma unroll
      for (int i = 0; i < 8; ++i) acc += (float)o8[i] * p8[i];
    }
    float logit = (lane < CC) ? acc * (1.0f / 0.07f) : -1e30f;
    float mx = wave_max(logit);
    float e = (lane < CC) ? __expf(logit - mx) : 0.f;
    float sum = wave_sum(e);
    float outv = (lane < CC) ? (e / sum) : (lane == CC ? 1.0f : 0.0f);
    scoreT[(((size_t)br * BB + b) * 32 + lane) * TT + t] = (_Float16)outv;
  }
}

// ---------------- flash-style refine: seg = mean_br softmax(o^T o) @ score ----------------
// |logit| <= 1 (normalized vectors) -> fixed max of 1, exp(l-1); denom via ones-column c=20.
// Block = 4 waves x 16 t-rows; the 32-row s-tile operand is async-staged into LDS
// (32 x 128K chunks, double-buffered) and shared by all 4 waves: 4x less L2 traffic.
__global__ void __launch_bounds__(128) k_attn(const _Float16* __restrict__ onorm,
                                              const _Float16* __restrict__ scoreT,
                                              float* __restrict__ dout) {
  __shared__ __align__(16) _Float16 stile[2][32 * 136];  // stride 136: bank-spread, 16B-aligned
  __shared__ __align__(16) _Float16 ptile[4][16 * 32];
  int tid = threadIdx.x;
  int lane = tid & 31, warp = tid >> 5;
  int blk = blockIdx.x;            // 128 = 4b * 32 t-blocks
  int b = blk >> 5;
  int t0 = (blk & 31) * 64 + warp * 16;
  int m = lane & 15, hi = lane >> 4;
  int j0 = tid >> 4, kp = tid & 15;   // staging: rows j0+{0,8,16,24}, 16B col kp
  float segA[8], segB[8];
#pragma unroll
  for (int r = 0; r < 8; ++r) { segA[r] = 0.f; segB[r] = 0.f; }
  _Float16* pt = &ptile[warp][0];
  for (int br = 0; br < 2; ++br) {
    const _Float16* base = onorm + ((size_t)b * TP + 1) * DD + br * D2;
    const _Float16* sT = scoreT + ((size_t)(br * BB + b)) * 32 * TT;
    const _Float16* arow = base + (size_t)t0 * DD;
    v8f r0 = zero8(), r1 = zero8();
    for (int st = 0; st < 64; ++st) {
      int s0 = st * 32;
      const _Float16* srcb = base + (size_t)(s0 + j0) * DD + kp * 8;
      _Float16* dstb = &stile[0][j0 * 136 + kp * 8];
      auto stage = [&](int c8, int buf) {        // 32 rows x 128 K halves (8KB)
        const _Float16* s = srcb + c8 * 128;
        _Float16* d = dstb + buf * (32 * 136);
        async_cp16(d, s);
        async_cp16(d + 8 * 136, s + (size_t)8 * DD);
        async_cp16(d + 16 * 136, s + (size_t)16 * DD);
        async_cp16(d + 24 * 136, s + (size_t)24 * DD);
      };
      v8f l0 = zero8(), l1 = zero8();
      stage(0, 0);
      wait_async0();
      __syncthreads();
      for (int c8 = 0; c8 < 8; ++c8) {
        int cur = c8 & 1;
        if (c8 < 7) stage(c8 + 1, cur ^ 1);
        int kb = c8 * 128;
#pragma unroll
        for (int kc = 0; kc < 4; ++kc) {
          int kloc = kc * 32;
          v16h a = load_A(arow + kb + kloc, DD, lane);
          v16h bf0 = load_B(&stile[cur][kloc], 136, lane);
          v16h bf1 = load_B(&stile[cur][16 * 136 + kloc], 136, lane);
          l0 = wmma16(a, bf0, l0);
          l1 = wmma16(a, bf1, l1);
        }
        if (c8 < 7) { wait_async0(); __syncthreads(); }
      }
      // exp + transpose via wave-private LDS tile, then P @ scoreT
#pragma unroll
      for (int r = 0; r < 8; ++r) {
        int row = r + 8 * hi;
        pt[row * 32 + m] = (_Float16)__expf(l0[r] - 1.0f);
        pt[row * 32 + 16 + m] = (_Float16)__expf(l1[r] - 1.0f);
      }
      lds_fence();
      v16h pa = load_A(pt, 32, lane);
      v16h sb0 = load_B(sT + s0, TT, lane);                   // classes 0..15
      v16h sb1 = load_B(sT + (size_t)16 * TT + s0, TT, lane); // classes 16..31 (20=ones)
      r0 = wmma16(pa, sb0, r0);
      r1 = wmma16(pa, sb1, r1);
      lds_fence();
    }
#pragma unroll
    for (int r = 0; r < 8; ++r) {
      float den = __shfl(r1[r], hi ? 20 : 4, 32);  // column N=4 of second tile == c=20
      den = fmaxf(den, 1e-30f);
      segA[r] += 0.5f * r0[r] / den;
      segB[r] += 0.5f * r1[r] / den;
    }
  }
  float* seg = dout + 160;
#pragma unroll
  for (int r = 0; r < 8; ++r) {
    int t = t0 + r + 8 * hi;
    seg[((size_t)b * TT + t) * CC + m] = segA[r];
    if (m < 4) seg[((size_t)b * TT + t) * CC + 16 + m] = segB[r];
  }
}

// ---------------- top-10 / bottom-10 means per (b,c) ----------------
__global__ void k_topk(const float* __restrict__ seg, float* __restrict__ outp) {
  int i = blockIdx.x * blockDim.x + threadIdx.x;
  if (i >= BB * CC) return;
  int b = i / CC, c = i % CC;
  float top[10], bot[10];
#pragma unroll
  for (int j = 0; j < 10; ++j) { top[j] = -1e30f; bot[j] = 1e30f; }
  for (int t = 0; t < TT; ++t) {
    float v = seg[((size_t)b * TT + t) * CC + c];
    int mi = 0; float mv = top[0];
#pragma unroll
    for (int j = 1; j < 10; ++j) if (top[j] < mv) { mv = top[j]; mi = j; }
    if (v > mv) top[mi] = v;
    int xi = 0; float xv = bot[0];
#pragma unroll
    for (int j = 1; j < 10; ++j) if (bot[j] > xv) { xv = bot[j]; xi = j; }
    if (v < xv) bot[xi] = v;
  }
  float as = 0.f, bs = 0.f;
#pragma unroll
  for (int j = 0; j < 10; ++j) { as += top[j]; bs += bot[j]; }
  outp[b * CC + c] = as * 0.1f;
  outp[80 + b * CC + c] = bs * 0.1f;
}

// ---------------- launcher ----------------
extern "C" void kernel_launch(void* const* d_in, const int* in_sizes, int n_in,
                              void* d_out, int out_size, void* d_ws, size_t ws_size,
                              hipStream_t stream) {
  (void)in_sizes; (void)n_in; (void)out_size; (void)ws_size;
  const float* x = (const float*)d_in[0];
  const float* w1 = (const float*)d_in[1];
  const float* b1 = (const float*)d_in[2];
  const float* w2 = (const float*)d_in[3];
  const float* b2 = (const float*)d_in[4];
  const float* proxy = (const float*)d_in[5];
  char* ws = (char*)d_ws;
  const size_t PBUF = (size_t)BB * TP * DD * 2;        // 33,587,200 B
  _Float16* bufA = (_Float16*)(ws + (size_t)0);        // xh, then h2
  _Float16* bufB = (_Float16*)(ws + PBUF);             // h1, then onorm
  _Float16* wp1  = (_Float16*)(ws + 2 * PBUF);
  _Float16* wp2  = (_Float16*)(ws + 2 * PBUF + 12582912);
  float*    pn   = (float*)   (ws + 2 * PBUF + 2 * (size_t)12582912);
  _Float16* scT  = (_Float16*)(ws + 2 * PBUF + 2 * (size_t)12582912 + 163840);
  float* dout = (float*)d_out;

  k_zeropad<<<16, 256, 0, stream>>>(bufA, bufB);
  k_cvt<<<8192, 256, 0, stream>>>(x, bufA);
  k_prepack<<<8192, 256, 0, stream>>>(w1, w2, wp1, wp2);
  k_proxy<<<40, 32, 0, stream>>>(proxy, pn);
  k_conv<<<2048, 128, 0, stream>>>(bufA, wp1, b1, bufB);   // conv1+ReLU
  k_conv<<<2048, 128, 0, stream>>>(bufB, wp2, b2, bufA);   // conv2+ReLU
  k_norm_feat<<<2048, 128, 0, stream>>>(bufA, pn, bufB, scT);
  k_attn<<<128, 128, 0, stream>>>(bufB, scT, dout);
  k_topk<<<1, 128, 0, stream>>>(dout + 160, dout);
}
